// SynthesisConstBlock_4483945857636
// MI455X (gfx1250) — matrix operand order repaired and checked by reading, not compiled, not run on hardware
//
#include <hip/hip_runtime.h>

// ---------------------------------------------------------------------------
// StyleGAN2 modulated conv, restructured for MI455X (gfx1250, wave32, WMMA):
//   out[b,o,p] = d[b,o] * conv3x3(cw*s_conv, style[b,:]*x[b,:])[o,p]
//                + bias[o] + noise_w[o]*noise[b,o,p]
// Demod factorizes -> the heavy conv uses ONE shared bf16 weight tensor
// (4.7 MB, L2-resident). Conv = implicit GEMM on v_wmma_f32_16x16x32_bf16,
// with an async global->LDS double-buffered pipeline (ASYNCcnt) feeding the
// B fragments via ds_load_b128 from a pre-zeroed halo tile.
// ---------------------------------------------------------------------------

#define B_    32
#define ICH   512
#define OCH   512
#define H_    64
#define W_    64

typedef __attribute__((ext_vector_type(16))) __bf16 v16bf;
typedef __attribute__((ext_vector_type(8)))  __bf16 v8bf;
typedef __attribute__((ext_vector_type(8)))  float  v8f;
typedef __attribute__((ext_vector_type(4)))  float  v4f;

// ---- CDNA5 async global->LDS copy (16B per lane), ASYNCcnt-tracked.
// Inline asm (portable across toolchains; the clang builtin's prototype
// differs between ROCm drops). vdst = 32-bit LDS byte address, addr = VGPR
// pair with the 64-bit global address, saddr = off.
__device__ __forceinline__ void async_cp16(const __bf16* g, __bf16* l) {
  unsigned lofs = (unsigned)(unsigned long long)(void*)l;   // LDS byte addr
  asm volatile("global_load_async_to_lds_b128 %0, %1, off"
               :: "v"(lofs), "v"((unsigned long long)g)
               : "memory");
}

#define WAIT_ASYNC(n) asm volatile("s_wait_asynccnt %0" :: "n"(n) : "memory")

// ---------------------------------------------------------------------------
// 1) style[b,i] = s_lin * sum_d w[b,d]*lin_w[i,d] + lin_b[i]
// ---------------------------------------------------------------------------
__global__ __launch_bounds__(256) void style_kernel(
    const float* __restrict__ w, const float* __restrict__ lin_w,
    const float* __restrict__ lin_b, float* __restrict__ style) {
  const float s_lin = 0.0625f; // sqrt(2/512)
  int idx = blockIdx.x * 256 + threadIdx.x;   // 32*512 threads
  int b = idx >> 9, i = idx & 511;
  const v4f* wp = (const v4f*)(w + (size_t)b * 512);
  const v4f* lp = (const v4f*)(lin_w + (size_t)i * 512);
  float acc = 0.f;
#pragma unroll 4
  for (int d = 0; d < 128; ++d) {
    v4f a = wp[d], c = lp[d];
    acc += a.x * c.x + a.y * c.y + a.z * c.z + a.w * c.w;
  }
  style[idx] = acc * s_lin + lin_b[i];
}

// ---------------------------------------------------------------------------
// 2) Weight prep: wT[tap][o][i] = bf16(cw[o][i][tap] * s_conv)   (K-contig!)
//    cw2sum[o][i] = sum_tap (cw*s_conv)^2
// ---------------------------------------------------------------------------
__global__ __launch_bounds__(256) void wprep_kernel(
    const float* __restrict__ conv_w, __bf16* __restrict__ wT,
    float* __restrict__ cw2sum) {
  const float s_conv = 0.02083333395f; // sqrt(2/(512*9))
  int idx = blockIdx.x * 256 + threadIdx.x;   // 512*512 threads
  int o = idx >> 9, i = idx & 511;
  size_t base = (size_t)idx * 9;
  float ss = 0.f;
#pragma unroll
  for (int t = 0; t < 9; ++t) {
    float v = conv_w[base + t] * s_conv;
    ss += v * v;
    wT[((size_t)t * OCH + o) * ICH + i] = (__bf16)v;
  }
  cw2sum[idx] = ss;
}

// ---------------------------------------------------------------------------
// 3) d[b,o] = rsqrt(sum_i style^2[b,i]*cw2sum[o,i] + 1e-8)
// ---------------------------------------------------------------------------
__global__ __launch_bounds__(256) void demod_kernel(
    const float* __restrict__ style, const float* __restrict__ cw2sum,
    float* __restrict__ dmod) {
  __shared__ float st2[ICH];
  int b = blockIdx.y;
  int o = blockIdx.x * 256 + threadIdx.x;
#pragma unroll
  for (int r = 0; r < 2; ++r) {
    int i = r * 256 + threadIdx.x;
    float sv = style[b * ICH + i];
    st2[i] = sv * sv;
  }
  __syncthreads();
  const v4f* cp = (const v4f*)(cw2sum + (size_t)o * ICH);
  float acc = 0.f;
#pragma unroll 4
  for (int i4 = 0; i4 < 128; ++i4) {
    v4f c = cp[i4];
    const float* s4 = &st2[i4 * 4];
    acc += c.x * s4[0] + c.y * s4[1] + c.z * s4[2] + c.w * s4[3];
  }
  dmod[b * OCH + o] = rsqrtf(acc + 1e-8f);
}

// ---------------------------------------------------------------------------
// 4) xmod[b][y][x][i] = bf16(style[b,i] * x[b,i,y,x])   (channels-last)
//    LDS tile transpose: 32 channels x 64 columns per block.
// ---------------------------------------------------------------------------
__global__ __launch_bounds__(256) void modx_kernel(
    const float* __restrict__ x, const float* __restrict__ style,
    __bf16* __restrict__ xmod) {
  __shared__ __align__(16) __bf16 tile[64][40];   // [x][ch], 80B rows
  int tid = threadIdx.x;
  int ic0 = blockIdx.x * 32;
  int yy  = blockIdx.y;
  int b   = blockIdx.z;
#pragma unroll
  for (int r = 0; r < 8; ++r) {
    int ch = r * 4 + (tid >> 6);
    int xx = tid & 63;
    float sv = style[b * ICH + ic0 + ch];
    float v  = x[(((size_t)b * ICH + ic0 + ch) * H_ + yy) * W_ + xx] * sv;
    tile[xx][ch] = (__bf16)v;
  }
  __syncthreads();
  int xx = tid >> 2;
  int cb = (tid & 3) * 8;
  v8bf v = *(const v8bf*)&tile[xx][cb];
  *(v8bf*)(xmod + (((size_t)b * H_ + yy) * W_ + xx) * ICH + ic0 + cb) = v;
}

// ---------------------------------------------------------------------------
// 5) Implicit-GEMM 3x3 conv with WMMA bf16 + async-LDS pipeline.
//    Grid: (32 spatial tiles of 2 rows, 4 M-tiles of 128 out-ch, 32 batch).
//    Block: 256 thr = 8 waves; wave (wm 0..3, wn 0..1) owns a 32x64 tile.
//    LDS tile per K-chunk: [4 rows][66 cols (halo)][32 ch] bf16 = 16.5 KB,
//    double-buffered; halo/OOB rows stay zero from one-time pre-clear.
// ---------------------------------------------------------------------------
#define LROW   (66 * 32)            // elements per LDS row
#define LBUF   (4 * LROW)           // elements per buffer (8448)

__global__ __launch_bounds__(256) void conv_wmma_kernel(
    const __bf16* __restrict__ xmod, const __bf16* __restrict__ wT,
    const float* __restrict__ dmod, const float* __restrict__ bias,
    const float* __restrict__ noise_w, const float* __restrict__ noise,
    float* __restrict__ out) {
  __shared__ __align__(16) __bf16 lbuf[2 * LBUF];   // 33,792 B

  int tid    = threadIdx.x;
  int lane   = tid & 31;
  int wave   = tid >> 5;
  int wm     = wave >> 1;        // 0..3 -> M offset
  int wn     = wave & 1;         // 0..1 -> row within spatial tile
  int laneN  = lane & 15;
  int laneHi = lane >> 4;

  int bz = blockIdx.z;
  int o0 = blockIdx.y * 128 + wm * 32;
  int y0 = blockIdx.x * 2;
  int y  = y0 + wn;

  // ---- one-time clear: halo cols / OOB rows stay zero for all 16 chunks
  unsigned* lz = (unsigned*)lbuf;
#pragma unroll 1
  for (int i = tid; i < (2 * LBUF) / 2; i += 256) lz[i] = 0u;
  __syncthreads();

  // ---- this thread's async-copy site: (row, col) of the 4x64 interior
  int  crow = tid >> 6;                         // 0..3 (uniform per wave)
  int  ccol = tid & 63;                         // 0..63
  int  giy  = y0 - 1 + crow;
  bool cok  = (giy >= 0) && (giy < H_);
  const __bf16* gsrc = xmod + (((size_t)bz * H_ + giy) * W_ + ccol) * ICH;
  __bf16* ldst = &lbuf[(crow * 66 + ccol + 1) * 32];        // +1: halo col

#define ISSUE_CHUNK(c)                                                    \
  do {                                                                    \
    if (cok) {                                                            \
      const __bf16* g_ = gsrc + (c) * 32;                                 \
      __bf16* l_ = ldst + ((c) & 1) * LBUF;                               \
      async_cp16(g_,      l_);                                            \
      async_cp16(g_ + 8,  l_ + 8);                                        \
      async_cp16(g_ + 16, l_ + 16);                                       \
      async_cp16(g_ + 24, l_ + 24);                                       \
    }                                                                     \
  } while (0)

  ISSUE_CHUNK(0);
  ISSUE_CHUNK(1);

  // ---- per-lane A base pointers (weights, K-contiguous per tap)
  const __bf16* paBase[2];
#pragma unroll
  for (int mf = 0; mf < 2; ++mf)
    paBase[mf] = wT + (size_t)(o0 + mf * 16 + laneN) * ICH + laneHi * 8;

  // ---- per-lane B base offset inside a buffer (tap/nf folds into ds imm)
  const int bofs = (wn * 66 + laneN) * 32 + laneHi * 8;

  v8f acc[2][4] = {{{}, {}, {}, {}}, {{}, {}, {}, {}}};

#pragma unroll 1
  for (int c = 0; c < 16; ++c) {
    if (c + 1 < 16) { WAIT_ASYNC(4); } else { WAIT_ASYNC(0); }
    __syncthreads();                       // chunk c resident for all waves

    const __bf16* lbase = &lbuf[(c & 1) * LBUF + bofs];
    const int ic0 = c * 32;

#pragma unroll
    for (int tap = 0; tap < 9; ++tap) {
      const int kh = tap / 3, kw = tap % 3;

      // A fragments: two b128 global loads per lane, documented 16-bit A
      // layout (lanes0-15: K{0..7,16..23}; lanes16-31: K{8..15,24..31}).
      v16bf afrag[2];
#pragma unroll
      for (int mf = 0; mf < 2; ++mf) {
        const __bf16* pa = paBase[mf] + (size_t)tap * (OCH * ICH) + ic0;
        v8bf a0 = *(const v8bf*)pa;
        v8bf a1 = *(const v8bf*)(pa + 16);
#pragma unroll
        for (int j = 0; j < 8; ++j) {
          afrag[mf][j]     = a0[j];
          afrag[mf][j + 8] = a1[j];
        }
      }

      // B fragments: two ds_load_b128 per lane from the halo tile
      // (lanes0-15: K{0..7,16..23}; lanes16-31: K{8..15,24..31}); no bounds checks needed.
#pragma unroll
      for (int nf = 0; nf < 4; ++nf) {
        const __bf16* pb = lbase + (kh * 66 + nf * 16 + kw) * 32;
        v8bf b0 = *(const v8bf*)pb;
        v8bf b1 = *(const v8bf*)(pb + 16);
        v16bf bfrag;
#pragma unroll
        for (int j = 0; j < 8; ++j) {
          bfrag[j]     = b0[j];
          bfrag[j + 8] = b1[j];
        }
#pragma unroll
        for (int mf = 0; mf < 2; ++mf) {
          acc[mf][nf] = __builtin_amdgcn_wmma_f32_16x16x32_bf16(
              false, afrag[mf], false, bfrag, (short)0, acc[mf][nf],
              false, false);
        }
      }
    }

    __syncthreads();                       // buf (c&1) free for rewrite
    if (c + 2 < 16) ISSUE_CHUNK(c + 2);
  }

  // ---- Epilogue: demod + bias + noise, fp32 out.
#pragma unroll
  for (int mf = 0; mf < 2; ++mf) {
    float dv[8], bb[8], nw[8];
#pragma unroll
    for (int r = 0; r < 8; ++r) {
      int o = o0 + mf * 16 + laneHi * 8 + r;
      dv[r] = dmod[bz * OCH + o];
      bb[r] = bias[o];
      nw[r] = noise_w[o];
    }
#pragma unroll
    for (int nf = 0; nf < 4; ++nf) {
      int col = nf * 16 + laneN;
#pragma unroll
      for (int r = 0; r < 8; ++r) {
        int o = o0 + mf * 16 + laneHi * 8 + r;
        size_t oidx = (((size_t)bz * OCH + o) * H_ + y) * W_ + col;
        out[oidx] = acc[mf][nf][r] * dv[r] + bb[r] + nw[r] * noise[oidx];
      }
    }
  }
}

// ---------------------------------------------------------------------------
extern "C" void kernel_launch(void* const* d_in, const int* in_sizes, int n_in,
                              void* d_out, int out_size, void* d_ws,
                              size_t ws_size, hipStream_t stream) {
  const float* x       = (const float*)d_in[0];
  const float* w       = (const float*)d_in[1];
  const float* conv_w  = (const float*)d_in[2];
  const float* lin_w   = (const float*)d_in[3];
  const float* lin_b   = (const float*)d_in[4];
  const float* bias    = (const float*)d_in[5];
  const float* noise_w = (const float*)d_in[6];
  const float* noise   = (const float*)d_in[7];
  float* out = (float*)d_out;

  // Workspace carve-up (~140.1 MB total, all 256B-aligned offsets).
  char* ws = (char*)d_ws;
  size_t off = 0;
  __bf16* xmod = (__bf16*)(ws + off); off += (size_t)B_ * H_ * W_ * ICH * 2;   // 134,217,728
  __bf16* wT   = (__bf16*)(ws + off); off += (size_t)9 * OCH * ICH * 2;        //   4,718,592
  float*  cw2  = (float*)(ws + off);  off += (size_t)OCH * ICH * 4;            //   1,048,576
  float*  styl = (float*)(ws + off);  off += (size_t)B_ * ICH * 4;             //      65,536
  float*  dmod = (float*)(ws + off);  off += (size_t)B_ * OCH * 4;             //      65,536

  style_kernel<<<dim3((B_ * ICH) / 256), 256, 0, stream>>>(w, lin_w, lin_b, styl);
  wprep_kernel<<<dim3((OCH * ICH) / 256), 256, 0, stream>>>(conv_w, wT, cw2);
  demod_kernel<<<dim3(OCH / 256, B_), 256, 0, stream>>>(styl, cw2, dmod);
  modx_kernel<<<dim3(ICH / 32, H_, B_), 256, 0, stream>>>(x, styl, xmod);
  conv_wmma_kernel<<<dim3(H_ / 2, OCH / 128, B_), 256, 0, stream>>>(
      xmod, wT, dmod, bias, noise_w, noise, out);
}
